// VLineHead_21509196218383
// MI455X (gfx1250) — matrix operand order repaired and checked by reading, not compiled, not run on hardware
//
#include <hip/hip_runtime.h>
#include <hip/hip_bf16.h>

// ---------------------------------------------------------------------------
// VLineHead pipeline for gfx1250 (CDNA5, wave32, WMMA).
// All dense contractions run on v_wmma_f32_16x16x32_f16 (f16 in, f32 accum)
// with 2x2 register blocking per wave (32x32 output patch / wave, 4 waves per
// 128-thread block). A tiles are staged into LDS with the gfx1250 async-copy
// path (global_load_async_to_lds_b128 + s_wait_asynccnt), double-buffered.
//
// WMMA fragment layouts (ISA 7.12.2, wave32):
//   A 16x32 f16 : lane<16 holds row M=lane,   K={0..7,16..23}
//                 lane>=16 holds row M=lane-16, K={8..15,24..31}
//   B 32x16 f16 : lane<16 col N=lane K={0..7,16..23}; lane>=16 col N=lane-16 K={8..15,24..31}
//   C/D 16x16 f32: lane -> N=lane&15, VGPR r -> M=r + (lane>=16 ? 8 : 0)
// B is read from [N,K] f16 (weights as stored) -> contiguous 32B runs.
// ---------------------------------------------------------------------------

typedef __attribute__((ext_vector_type(16))) _Float16 v16h;
typedef __attribute__((ext_vector_type(8)))  float    v8f;

#define DEVI static __device__ __forceinline__
#define ALD 40   // LDS A-tile row pitch in halfs (80B: 16B-aligned, bank-spread)

DEVI v16h load_a16(const _Float16* __restrict__ p, int k0) {
  // p = row base + (lane>=16 ? 8 : 0); two contiguous 16B runs
  v16h a;
#pragma unroll
  for (int e = 0; e < 8; ++e) { a[e] = p[k0 + e]; a[e + 8] = p[k0 + 16 + e]; }
  return a;
}
DEVI v16h load_b16(const _Float16* __restrict__ p, int k0) {
  // p = col base + (lane>=16 ? 8 : 0); two contiguous 16B runs
  v16h b;
#pragma unroll
  for (int e = 0; e < 8; ++e) { b[e] = p[k0 + e]; b[e + 8] = p[k0 + 16 + e]; }
  return b;
}
DEVI v16h lds_a_frag(const _Float16* p) {
  v16h a;
#pragma unroll
  for (int e = 0; e < 8; ++e) { a[e] = p[e]; a[e + 8] = p[16 + e]; }
  return a;
}
DEVI v8f wmma_step(v16h a, v16h b, v8f c) {
  return __builtin_amdgcn_wmma_f32_16x16x32_f16(false, a, false, b, (short)0, c,
                                                false, false);
}
// gfx1250 async copy: 16B global -> LDS, tracked by ASYNCcnt
DEVI void stage_a_async(const _Float16* __restrict__ gsrc, _Float16* lds_dst) {
  asm volatile("global_load_async_to_lds_b128 %0, %1, off"
               :: "v"((unsigned)(unsigned long long)lds_dst),
                  "v"((unsigned long long)gsrc)
               : "memory");
}
DEVI void wait_async_all() { asm volatile("s_wait_asynccnt 0x0" ::: "memory"); }

DEVI void store_tile(v8f acc, _Float16* __restrict__ C, int row, int col, int ldc,
                     float bb, int relu) {
#pragma unroll
  for (int r = 0; r < 8; ++r) {
    float v = acc[r] + bb;
    if (relu) v = fmaxf(v, 0.0f);
    C[(size_t)(row + r) * ldc + col] = (_Float16)v;
  }
}

// ---------------- generic GEMM: C = A @ W^T + bias, 32x128 per block --------
// A: f16 [M,K]; W: f16 [N,K]; C: f16 [M,N].  M%32==0, N%128==0, K%32==0.
__global__ void __launch_bounds__(128)
k_gemm32(const _Float16* __restrict__ A, const _Float16* __restrict__ W,
         const float* __restrict__ bias, _Float16* __restrict__ C,
         int M, int Nn, int K, int relu) {
  __shared__ _Float16 smA[2][32 * ALD];
  const int tid = threadIdx.x, lane = tid & 31, wave = tid >> 5;
  const int l15 = lane & 15, hi = lane >> 4;
  const int row0 = blockIdx.x * 32;
  const int col0 = blockIdx.y * 128 + wave * 32;
  // A staging: thread t moves 16B of row t/4
  const int srow = tid >> 2, sseg = (tid & 3) * 8;
  const _Float16* asrc = A + (size_t)(row0 + srow) * K + sseg;
  _Float16* sd0 = &smA[0][srow * ALD + sseg];
  _Float16* sd1 = &smA[1][srow * ALD + sseg];
  const _Float16* b0 = W + (size_t)(col0 + l15) * K + hi * 8;
  const _Float16* b1 = b0 + (size_t)16 * K;
  v8f a00 = {}, a01 = {}, a10 = {}, a11 = {};
  stage_a_async(asrc, sd0);
  wait_async_all();
  __syncthreads();
  int buf = 0;
  for (int k0 = 0; k0 < K; k0 += 32) {
    if (k0 + 32 < K) {
      stage_a_async(asrc + k0 + 32, buf ? sd0 : sd1);
      __builtin_prefetch(b0 + k0 + 64, 0, 1);
      __builtin_prefetch(b1 + k0 + 64, 0, 1);
    }
    const _Float16* at = &smA[buf][0] + l15 * ALD + hi * 8;
    v16h fa0 = lds_a_frag(at);
    v16h fa1 = lds_a_frag(at + 16 * ALD);
    v16h fb0 = load_b16(b0, k0);
    v16h fb1 = load_b16(b1, k0);
    a00 = wmma_step(fa0, fb0, a00);
    a01 = wmma_step(fa0, fb1, a01);
    a10 = wmma_step(fa1, fb0, a10);
    a11 = wmma_step(fa1, fb1, a11);
    wait_async_all();
    __syncthreads();
    buf ^= 1;
  }
  const int ccol = col0 + l15;
  const int crow = row0 + hi * 8;
  const float bb0 = bias ? bias[ccol] : 0.0f;
  const float bb1 = bias ? bias[ccol + 16] : 0.0f;
  store_tile(a00, C, crow,      ccol,      Nn, bb0, relu);
  store_tile(a01, C, crow,      ccol + 16, Nn, bb1, relu);
  store_tile(a10, C, crow + 16, ccol,      Nn, bb0, relu);
  store_tile(a11, C, crow + 16, ccol + 16, Nn, bb1, relu);
}

// -------- direct 3x3 SAME conv as 9 shifted GEMMs (K = 9*256 = 2304) --------
// rT: f16 [64][24][24][256] channel-last; wT: f16 [9][128][256];
// fC: f16 [64][128][576] = relu(conv+bias).  32 positions x 128 channels/block.
__global__ void __launch_bounds__(128)
k_conv3x3(const _Float16* __restrict__ rT, const _Float16* __restrict__ wT,
          const float* __restrict__ bias, _Float16* __restrict__ fC) {
  const int tid = threadIdx.x, lane = tid & 31, wave = tid >> 5;
  const int l15 = lane & 15, hi = lane >> 4;
  const int p0 = blockIdx.x * 32;        // 32 consecutive positions, one image
  const int o0 = wave * 32;              // 32 output channels per wave
  const int n = p0 / 576;
  const int hwA = (p0 + l15) % 576,      yA = hwA / 24, xA = hwA % 24;
  const int hwB = (p0 + 16 + l15) % 576, yB = hwB / 24, xB = hwB % 24;
  const _Float16* rbase = rT + (size_t)n * (576 * 256) + hi * 8;
  const _Float16* b0 = wT + (size_t)(o0 + l15) * 256 + hi * 8;
  const _Float16* b1 = b0 + 16 * 256;
  v8f a00 = {}, a01 = {}, a10 = {}, a11 = {};
#pragma unroll
  for (int kh = 0; kh < 3; ++kh) {
#pragma unroll
    for (int kw = 0; kw < 3; ++kw) {
      const int yyA = yA + kh - 1, xxA = xA + kw - 1;
      const int yyB = yB + kh - 1, xxB = xB + kw - 1;
      const bool okA = (yyA >= 0) & (yyA < 24) & (xxA >= 0) & (xxA < 24);
      const bool okB = (yyB >= 0) & (yyB < 24) & (xxB >= 0) & (xxB < 24);
      const _Float16* aA = rbase + (size_t)((yyA * 24 + xxA) * 256);
      const _Float16* aB = rbase + (size_t)((yyB * 24 + xxB) * 256);
      const _Float16* bk0 = b0 + (size_t)(kh * 3 + kw) * (128 * 256);
      const _Float16* bk1 = b1 + (size_t)(kh * 3 + kw) * (128 * 256);
      for (int k0 = 0; k0 < 256; k0 += 32) {
        v16h fa0 = {}, fa1 = {};           // zero rows implement SAME padding
        if (okA) fa0 = load_a16(aA, k0);
        if (okB) fa1 = load_a16(aB, k0);
        v16h fb0 = load_b16(bk0, k0);
        v16h fb1 = load_b16(bk1, k0);
        a00 = wmma_step(fa0, fb0, a00);
        a01 = wmma_step(fa0, fb1, a01);
        a10 = wmma_step(fa1, fb0, a10);
        a11 = wmma_step(fa1, fb1, a11);
      }
    }
  }
  const float bb0 = bias[o0 + l15], bb1 = bias[o0 + 16 + l15];
  _Float16* outn = fC + (size_t)n * (128 * 576);
#pragma unroll
  for (int r = 0; r < 8; ++r) {
    const int pa = (p0 + r + hi * 8) % 576;
    const int pb = (p0 + 16 + r + hi * 8) % 576;
    outn[(size_t)(o0 + l15) * 576 + pa]      = (_Float16)fmaxf(a00[r] + bb0, 0.f);
    outn[(size_t)(o0 + 16 + l15) * 576 + pa] = (_Float16)fmaxf(a01[r] + bb1, 0.f);
    outn[(size_t)(o0 + l15) * 576 + pb]      = (_Float16)fmaxf(a10[r] + bb0, 0.f);
    outn[(size_t)(o0 + 16 + l15) * 576 + pb] = (_Float16)fmaxf(a11[r] + bb1, 0.f);
  }
}

// -------- group mean-pool as per-image GEMM: [32x576]x[576x128], 1 block/n --
__global__ void __launch_bounds__(128)
k_poolgemm32(const _Float16* __restrict__ mT, const _Float16* __restrict__ fC,
             const float* __restrict__ counts, _Float16* __restrict__ outp) {
  __shared__ _Float16 smA[2][32 * ALD];
  const int tid = threadIdx.x, lane = tid & 31, wave = tid >> 5;
  const int l15 = lane & 15, hi = lane >> 4;
  const int n = blockIdx.z;
  const int col0 = wave * 32;
  const int srow = tid >> 2, sseg = (tid & 3) * 8;
  const _Float16* asrc = mT + (size_t)n * (32 * 576) + (size_t)srow * 576 + sseg;
  _Float16* sd0 = &smA[0][srow * ALD + sseg];
  _Float16* sd1 = &smA[1][srow * ALD + sseg];
  const _Float16* b0 = fC + (size_t)n * (128 * 576) + (size_t)(col0 + l15) * 576 + hi * 8;
  const _Float16* b1 = b0 + (size_t)16 * 576;
  v8f a00 = {}, a01 = {}, a10 = {}, a11 = {};
  stage_a_async(asrc, sd0);
  wait_async_all();
  __syncthreads();
  int buf = 0;
  for (int k0 = 0; k0 < 576; k0 += 32) {
    if (k0 + 32 < 576) stage_a_async(asrc + k0 + 32, buf ? sd0 : sd1);
    const _Float16* at = &smA[buf][0] + l15 * ALD + hi * 8;
    v16h fa0 = lds_a_frag(at);
    v16h fa1 = lds_a_frag(at + 16 * ALD);
    v16h fb0 = load_b16(b0, k0);
    v16h fb1 = load_b16(b1, k0);
    a00 = wmma_step(fa0, fb0, a00);
    a01 = wmma_step(fa0, fb1, a01);
    a10 = wmma_step(fa1, fb0, a10);
    a11 = wmma_step(fa1, fb1, a11);
    wait_async_all();
    __syncthreads();
    buf ^= 1;
  }
  const int c0 = col0 + l15;
  _Float16* outn = outp + (size_t)n * (32 * 128);
  const float* cn = counts + n * 32;
#pragma unroll
  for (int r = 0; r < 8; ++r) {
    const int va = r + hi * 8, vb = va + 16;
    outn[(size_t)va * 128 + c0]      = (_Float16)(a00[r] / cn[va]);
    outn[(size_t)va * 128 + c0 + 16] = (_Float16)(a01[r] / cn[va]);
    outn[(size_t)vb * 128 + c0]      = (_Float16)(a10[r] / cn[vb]);
    outn[(size_t)vb * 128 + c0 + 16] = (_Float16)(a11[r] / cn[vb]);
  }
}

// --------------------------- glue / prep kernels ----------------------------
__global__ void k_f32_to_f16(const float* __restrict__ s, _Float16* __restrict__ d,
                             int nElem) {
  int i = blockIdx.x * blockDim.x + threadIdx.x;
  if (i < nElem) d[i] = (_Float16)s[i];
}

__global__ void k_roi_T(const float* __restrict__ roi, _Float16* __restrict__ rT) {
  int i = blockIdx.x * blockDim.x + threadIdx.x;
  if (i >= 64 * 576 * 256) return;
  int c = i & 255;
  int t = i >> 8;
  int hw = t % 576, n = t / 576;
  rT[i] = (_Float16)roi[((size_t)n * 256 + c) * 576 + hw];
}

__global__ void k_convw_T(const float* __restrict__ w, _Float16* __restrict__ wT) {
  int i = blockIdx.x * blockDim.x + threadIdx.x;
  if (i >= 9 * 128 * 256) return;
  int c = i & 255;
  int t = i >> 8;
  int o = t % 128, kk = t / 128;
  wT[i] = (_Float16)w[((size_t)o * 256 + c) * 9 + kk];
}

__global__ void k_gpool(const float* __restrict__ roi, _Float16* __restrict__ gp) {
  int i = blockIdx.x * blockDim.x + threadIdx.x;
  if (i >= 64 * 2304) return;
  int d = i % 2304, n = i / 2304;
  int c = d / 9, g = d % 9, gy = g / 3, gx = g % 3;
  const float* base = roi + ((size_t)n * 256 + c) * 576 + gy * 8 * 24 + gx * 8;
  float s = 0.f;
#pragma unroll
  for (int yy = 0; yy < 8; ++yy)
#pragma unroll
    for (int xx = 0; xx < 8; ++xx) s += base[yy * 24 + xx];
  gp[i] = (_Float16)(s * (1.0f / 64.0f));
}

__global__ void k_maps16(const int* __restrict__ maps, _Float16* __restrict__ mT) {
  int i = blockIdx.x * blockDim.x + threadIdx.x;
  if (i >= 64 * 32 * 576) return;
  int hw = i % 576;
  int t = i / 576;
  int v = t & 31, n = t >> 5;
  mT[i] = (_Float16)(float)maps[((size_t)n * 576 + hw) * 32 + v];
}

__global__ void k_maxpool(const int* __restrict__ maps, const _Float16* __restrict__ fC,
                          _Float16* __restrict__ outp) {
  int i = blockIdx.x * blockDim.x + threadIdx.x;
  if (i >= 64 * 32 * 128) return;
  int c = i & 127;
  int t = i >> 7;
  int v = t & 31, n = t >> 5;
  const int* mp = maps + (size_t)n * (576 * 32) + v;
  const _Float16* f = fC + ((size_t)n * 128 + c) * 576;
  float mx = -1e30f;
  for (int hw = 0; hw < 576; ++hw)
    if (mp[(size_t)hw * 32] > 0) mx = fmaxf(mx, (float)f[hw]);
  outp[i] = (_Float16)mx;   // counts >= 1 always in reference
}

__global__ void k_vreduce(const _Float16* __restrict__ vf, const _Float16* __restrict__ vfm,
                          _Float16* __restrict__ pmn, _Float16* __restrict__ pmx) {
  int i = blockIdx.x * blockDim.x + threadIdx.x;
  if (i >= 64 * 512) return;
  int fl = i & 511, n = i >> 9;
  const _Float16* a = vf  + (size_t)(n * 32) * 512 + fl;
  const _Float16* b = vfm + (size_t)(n * 32) * 512 + fl;
  float s = 0.f, mx = -1e30f;
  for (int v = 0; v < 32; ++v) {
    s += (float)a[(size_t)v * 512];
    mx = fmaxf(mx, (float)b[(size_t)v * 512]);
  }
  pmn[i] = (_Float16)(s * (1.0f / 32.0f));
  pmx[i] = (_Float16)mx;
}

// all = [eye(32) | vf(512) | vfm(512) | pmn(512) | pmx(512) | g(256)]  (D=2336)
__global__ void k_concat(const _Float16* __restrict__ vf, const _Float16* __restrict__ vfm,
                         const _Float16* __restrict__ pmn, const _Float16* __restrict__ pmx,
                         const _Float16* __restrict__ g16, _Float16* __restrict__ allb) {
  int i = blockIdx.x * blockDim.x + threadIdx.x;
  if (i >= 2048 * 2336) return;
  int d = i % 2336, m = i / 2336;
  int v = m & 31, n = m >> 5;
  _Float16 val;
  if (d < 32)        val = (_Float16)(d == v ? 1.0f : 0.0f);
  else if (d < 544)  val = vf [(size_t)m * 512 + (d - 32)];
  else if (d < 1056) val = vfm[(size_t)m * 512 + (d - 544)];
  else if (d < 1568) val = pmn[(size_t)n * 512 + (d - 1056)];
  else if (d < 2080) val = pmx[(size_t)n * 512 + (d - 1568)];
  else               val = g16[(size_t)n * 256 + (d - 2080)];
  allb[i] = val;
}

// out[m,j] = hid[m,:] . Wa[j,:] + ba[j]   (N=2 -> plain VALU dot)
__global__ void k_head_final(const _Float16* __restrict__ hid, const float* __restrict__ Wa,
                             const float* __restrict__ ba, float* __restrict__ outp) {
  int i = blockIdx.x * blockDim.x + threadIdx.x;
  if (i >= 2048 * 2) return;
  int j = i & 1, m = i >> 1;
  const _Float16* h = hid + (size_t)m * 512;
  const float* w = Wa + (size_t)j * 512;
  float s = ba[j];
  for (int k = 0; k < 512; ++k) s += (float)h[k] * w[k];
  outp[i] = s;
}

// small heads on concat [eye|vf|pm|g] (D2=1312), composed on the fly
__global__ void k_head_small(const _Float16* __restrict__ vf, const _Float16* __restrict__ pm,
                             const _Float16* __restrict__ g16, const float* __restrict__ W,
                             const float* __restrict__ b, float* __restrict__ outp) {
  int i = blockIdx.x * blockDim.x + threadIdx.x;
  if (i >= 2048 * 2) return;
  int j = i & 1, m = i >> 1;
  int v = m & 31, n = m >> 5;
  const float* w = W + (size_t)j * 1312;
  float s = b[j] + w[v];
  const _Float16* a = vf + (size_t)m * 512;
  for (int k = 0; k < 512; ++k) s += (float)a[k] * w[32 + k];
  const _Float16* p = pm + (size_t)n * 512;
  for (int k = 0; k < 512; ++k) s += (float)p[k] * w[544 + k];
  const _Float16* gg = g16 + (size_t)n * 256;
  for (int k = 0; k < 256; ++k) s += (float)gg[k] * w[1056 + k];
  outp[i] = s;
}

// ---------------------------------------------------------------------------
extern "C" void kernel_launch(void* const* d_in, const int* in_sizes, int n_in,
                              void* d_out, int out_size, void* d_ws, size_t ws_size,
                              hipStream_t stream) {
  (void)in_sizes; (void)n_in; (void)out_size; (void)ws_size;
  const float* roi    = (const float*)d_in[0];
  const int*   maps_h = (const int*)d_in[1];
  const int*   maps_v = (const int*)d_in[2];
  const float* cnt_h  = (const float*)d_in[3];
  const float* cnt_v  = (const float*)d_in[4];
  const float* conv_w = (const float*)d_in[5];
  const float* conv_b = (const float*)d_in[6];
  const float* Wg  = (const float*)d_in[7];   const float* bg   = (const float*)d_in[8];
  const float* Wl  = (const float*)d_in[9];   const float* bl   = (const float*)d_in[10];
  const float* Wlv = (const float*)d_in[11];  const float* blv  = (const float*)d_in[12];
  const float* Wlm = (const float*)d_in[13];  const float* blm  = (const float*)d_in[14];
  const float* Wlvm= (const float*)d_in[15];  const float* blvm = (const float*)d_in[16];
  const float* Wc  = (const float*)d_in[17];  const float* bc   = (const float*)d_in[18];
  const float* Wcv = (const float*)d_in[19];  const float* bcv  = (const float*)d_in[20];
  const float* Wca = (const float*)d_in[21];  const float* bca  = (const float*)d_in[22];
  const float* Wcva= (const float*)d_in[23];  const float* bcva = (const float*)d_in[24];
  const float* Wcm = (const float*)d_in[25];  const float* bcm  = (const float*)d_in[26];
  const float* Wcmv= (const float*)d_in[27];  const float* bcmv = (const float*)d_in[28];
  const float* Wcx = (const float*)d_in[29];  const float* bcx  = (const float*)d_in[30];
  const float* Wcxv= (const float*)d_in[31];  const float* bcxv = (const float*)d_in[32];
  float* out = (float*)d_out;

  // workspace layout (~75 MB, 256B-aligned segments, all f16)
  char* ws = (char*)d_ws;
  size_t off = 0;
  auto alloc = [&](size_t bytes) -> char* {
    char* p = ws + off;
    off = (off + bytes + 255) & ~(size_t)255;
    return p;
  };
  _Float16* rT    = (_Float16*)alloc(64ull*576*256*2);
  _Float16* wT    = (_Float16*)alloc(9ull*128*256*2);
  _Float16* gp    = (_Float16*)alloc(64ull*2304*2);
  _Float16* Wg16  = (_Float16*)alloc(256ull*2304*2);
  _Float16* g16   = (_Float16*)alloc(64ull*256*2);
  _Float16* fC    = (_Float16*)alloc(64ull*128*576*2);
  _Float16* mTh   = (_Float16*)alloc(64ull*32*576*2);
  _Float16* mTv   = (_Float16*)alloc(64ull*32*576*2);
  _Float16* mph   = (_Float16*)alloc(64ull*32*128*2);
  _Float16* mpv   = (_Float16*)alloc(64ull*32*128*2);
  _Float16* mxh   = (_Float16*)alloc(64ull*32*128*2);
  _Float16* mxv   = (_Float16*)alloc(64ull*32*128*2);
  _Float16* Wl16  = (_Float16*)alloc(512ull*128*2);
  _Float16* Wlv16 = (_Float16*)alloc(512ull*128*2);
  _Float16* Wlm16 = (_Float16*)alloc(512ull*128*2);
  _Float16* Wlvm16= (_Float16*)alloc(512ull*128*2);
  _Float16* vf    = (_Float16*)alloc(2048ull*512*2);
  _Float16* vfv   = (_Float16*)alloc(2048ull*512*2);
  _Float16* vfm   = (_Float16*)alloc(2048ull*512*2);
  _Float16* vfvm  = (_Float16*)alloc(2048ull*512*2);
  _Float16* pmn   = (_Float16*)alloc(64ull*512*2);
  _Float16* pmx   = (_Float16*)alloc(64ull*512*2);
  _Float16* pmnv  = (_Float16*)alloc(64ull*512*2);
  _Float16* pmxv  = (_Float16*)alloc(64ull*512*2);
  _Float16* Wc16  = (_Float16*)alloc(512ull*2336*2);
  _Float16* Wcv16 = (_Float16*)alloc(512ull*2336*2);
  _Float16* allh  = (_Float16*)alloc(2048ull*2336*2);
  _Float16* allv  = (_Float16*)alloc(2048ull*2336*2);
  _Float16* hidh  = (_Float16*)alloc(2048ull*512*2);
  _Float16* hidv  = (_Float16*)alloc(2048ull*512*2);

  const int TB = 256;
  auto nb = [](long long n, int tb) { return (unsigned)((n + tb - 1) / tb); };

  // --- prep & f16 conversions ---
  k_roi_T   <<<nb(64ll*576*256, TB), TB, 0, stream>>>(roi, rT);
  k_convw_T <<<nb(9ll*128*256, TB), TB, 0, stream>>>(conv_w, wT);
  k_gpool   <<<nb(64ll*2304, TB), TB, 0, stream>>>(roi, gp);
  k_maps16  <<<nb(64ll*32*576, TB), TB, 0, stream>>>(maps_h, mTh);
  k_maps16  <<<nb(64ll*32*576, TB), TB, 0, stream>>>(maps_v, mTv);
  k_f32_to_f16<<<nb(256ll*2304, TB), TB, 0, stream>>>(Wg, Wg16, 256*2304);
  k_f32_to_f16<<<nb(512ll*128, TB), TB, 0, stream>>>(Wl, Wl16, 512*128);
  k_f32_to_f16<<<nb(512ll*128, TB), TB, 0, stream>>>(Wlv, Wlv16, 512*128);
  k_f32_to_f16<<<nb(512ll*128, TB), TB, 0, stream>>>(Wlm, Wlm16, 512*128);
  k_f32_to_f16<<<nb(512ll*128, TB), TB, 0, stream>>>(Wlvm, Wlvm16, 512*128);
  k_f32_to_f16<<<nb(512ll*2336, TB), TB, 0, stream>>>(Wc, Wc16, 512*2336);
  k_f32_to_f16<<<nb(512ll*2336, TB), TB, 0, stream>>>(Wcv, Wcv16, 512*2336);

  // --- global branch: g = relu(gp @ Wg^T + bg), [64,256] ---
  k_gemm32<<<dim3(2, 2), 128, 0, stream>>>(gp, Wg16, bg, g16, 64, 256, 2304, 1);

  // --- conv3x3 + bias + relu -> fC [64,128,576] ---
  k_conv3x3<<<dim3(1152), 128, 0, stream>>>(rT, wT, conv_b, fC);

  // --- vline pools ---
  k_poolgemm32<<<dim3(1, 1, 64), 128, 0, stream>>>(mTh, fC, cnt_h, mph);
  k_poolgemm32<<<dim3(1, 1, 64), 128, 0, stream>>>(mTv, fC, cnt_v, mpv);
  k_maxpool<<<nb(64ll*32*128, TB), TB, 0, stream>>>(maps_h, fC, mxh);
  k_maxpool<<<nb(64ll*32*128, TB), TB, 0, stream>>>(maps_v, fC, mxv);

  // --- per-branch first linears [2048,128]x[128,512] ---
  k_gemm32<<<dim3(64, 4), 128, 0, stream>>>(mph, Wl16,   bl,   vf,   2048, 512, 128, 0);
  k_gemm32<<<dim3(64, 4), 128, 0, stream>>>(mpv, Wlv16,  blv,  vfv,  2048, 512, 128, 0);
  k_gemm32<<<dim3(64, 4), 128, 0, stream>>>(mxh, Wlm16,  blm,  vfm,  2048, 512, 128, 0);
  k_gemm32<<<dim3(64, 4), 128, 0, stream>>>(mxv, Wlvm16, blvm, vfvm, 2048, 512, 128, 0);

  // --- mean/max over V ---
  k_vreduce<<<nb(64ll*512, TB), TB, 0, stream>>>(vf,  vfm,  pmn,  pmx);
  k_vreduce<<<nb(64ll*512, TB), TB, 0, stream>>>(vfv, vfvm, pmnv, pmxv);

  // --- concat features [2048,2336] ---
  k_concat<<<nb(2048ll*2336, TB), TB, 0, stream>>>(vf,  vfm,  pmn,  pmx,  g16, allh);
  k_concat<<<nb(2048ll*2336, TB), TB, 0, stream>>>(vfv, vfvm, pmnv, pmxv, g16, allv);

  // --- big heads: relu(all @ Wc^T + bc) [2048,2336]x[2336,512] ---
  k_gemm32<<<dim3(64, 4), 128, 0, stream>>>(allh, Wc16,  bc,  hidh, 2048, 512, 2336, 1);
  k_gemm32<<<dim3(64, 4), 128, 0, stream>>>(allv, Wcv16, bcv, hidv, 2048, 512, 2336, 1);

  // --- output heads (N=2): out layout [6][64][32][2] ---
  k_head_final<<<nb(4096, TB), TB, 0, stream>>>(hidh, Wca,  bca,  out + 0);
  k_head_final<<<nb(4096, TB), TB, 0, stream>>>(hidv, Wcva, bcva, out + 4096);
  k_head_small<<<nb(4096, TB), TB, 0, stream>>>(vf,   pmn,  g16, Wcm,  bcm,  out + 8192);
  k_head_small<<<nb(4096, TB), TB, 0, stream>>>(vfv,  pmnv, g16, Wcmv, bcmv, out + 12288);
  k_head_small<<<nb(4096, TB), TB, 0, stream>>>(vfm,  pmx,  g16, Wcx,  bcx,  out + 16384);
  k_head_small<<<nb(4096, TB), TB, 0, stream>>>(vfvm, pmxv, g16, Wcxv, bcxv, out + 20480);
}
